// Projector_8632884265563
// MI455X (gfx1250) — matrix operand order repaired and checked by reading, not compiled, not run on hardware
//
#include <hip/hip_runtime.h>
#include <math.h>

// ---- problem constants (match reference) ----
#define BS      8
#define N_PTS   8192
#define GRID_HW 64
#define KC      32                 // points per chunk = K of one WMMA
#define NCHUNK  (N_PTS / KC)       // 256
#define NSPLIT  8                  // blocks per batch along N
#define CH_PER  (NCHUNK / NSPLIT)  // 32 chunks per block
#define THREADS 512                // 16 wave32s -> 16 output tiles of 16x16
#define OUT_ELEMS (BS * GRID_HW * GRID_HW)

typedef __attribute__((ext_vector_type(16))) _Float16 v16h;
typedef __attribute__((ext_vector_type(8)))  _Float16 v8h;
typedef __attribute__((ext_vector_type(8)))  float    v8f;

__global__ __launch_bounds__(THREADS)
void proj_zero_kernel(float* __restrict__ out) {
    int i = blockIdx.x * blockDim.x + threadIdx.x;
    if (i < OUT_ELEMS) out[i] = 0.0f;
}

__global__ __launch_bounds__(THREADS)
void proj_splat_wmma_kernel(const float* __restrict__ xyz,
                            const float* __restrict__ az,
                            const float* __restrict__ el,
                            float* __restrict__ out)
{
    // LDS tiles: row = grid coordinate (h or w), 32 k-values contiguous per row.
    __shared__ _Float16 gxs[GRID_HW * KC];   // 4 KB: gx chunk, A-matrix (h-major)
    __shared__ _Float16 gys[GRID_HW * KC];   // 4 KB: gy chunk, B-matrix (w-major)
    __shared__ float    xs[KC];
    __shared__ float    ys[KC];

    const int b     = blockIdx.x;        // batch
    const int split = blockIdx.y;        // N-split
    const int tid   = threadIdx.x;
    const int lane  = tid & 31;
    const int wid   = tid >> 5;          // 0..15 -> one 16x16 output tile
    const int th    = wid >> 2;          // tile row (H)
    const int tw    = wid & 3;           // tile col (W)
    const int m     = lane & 15;
    const int hi    = lane >> 4;         // half-wave selector

    // ---- camera rotation R = Rel * Raz (uniform across block) ----
    const float a = az[b], e = el[b];
    const float ca = __cosf(a), sa = __sinf(a);
    const float ce = __cosf(e), se = __sinf(e);
    const float r00 = ce * ca,  r01 = ce * sa,  r02 = se;
    const float r10 = -sa,      r11 = ca;                    // r12 = 0
    const float r20 = -se * ca, r21 = -se * sa, r22 = ce;

    const float alpha = 120.0f;   // 60 * 64 / 32
    const float u0    = 32.0f;

    v8f acc = {};
    const float* pbase = xyz + (size_t)b * N_PTS * 3;

    const int c0 = split * CH_PER;
    for (int c = c0; c < c0 + CH_PER; ++c) {
        __syncthreads();                       // previous chunk fully consumed
        if (tid < KC) {                        // project 32 points of this chunk
            const int n = c * KC + tid;
            const float X = pbase[n * 3 + 0];
            const float Y = pbase[n * 3 + 1];
            const float Z = pbase[n * 3 + 2];
            const float Xc = r00 * X + r01 * Y + r02 * Z;
            const float Yc = r10 * X + r11 * Y;
            const float Zc = r20 * X + r21 * Y + r22 * Z - 2.0f;
            const float inv_az = 1.0f / fabsf(Zc);
            xs[tid] = (alpha * Xc - u0 * Zc) * inv_az;
            ys[tid] = (alpha * Yc - u0 * Zc) * inv_az;
        }
        __syncthreads();                       // xs/ys ready
        // Fill 2x 2048 f16 entries: thread -> 4 entries of each tile.
        #pragma unroll
        for (int j = 0; j < 4; ++j) {
            const int ei = tid + THREADS * j;  // flat = coord*32 + k
            const int g  = ei >> 5;            // grid coordinate (h for gx, w for gy)
            const int kk = ei & 31;            // point within chunk
            const float dx = xs[kk] - (float)g;
            const float dy = ys[kk] - (float)g;
            gxs[ei] = (_Float16)__expf(-dx * dx);   // inv2s == 1.0
            gys[ei] = (_Float16)__expf(-dy * dy);
        }
        __syncthreads();                       // tiles ready for WMMA

        // A fragment (16x32 f16): row h = 16*th + m,
        // lanes 0-15: K in {0..7, 16..23}; lanes 16-31: +8.  Two ds_load_b128.
        const int koff = hi * 8;
        const _Float16* arow = &gxs[(th * 16 + m) * KC];
        v8h a_lo = *reinterpret_cast<const v8h*>(arow + koff);
        v8h a_hi = *reinterpret_cast<const v8h*>(arow + 16 + koff);
        v16h afrag = __builtin_shufflevector(a_lo, a_hi,
            0, 1, 2, 3, 4, 5, 6, 7, 8, 9, 10, 11, 12, 13, 14, 15);

        // B fragment (32x16 f16): col w = 16*tw + m,
        // lanes 0-15: K in {0..7, 16..23}; lanes 16-31: +8.  Two ds_load_b128.
        const _Float16* brow = &gys[(tw * 16 + m) * KC + hi * 8];
        v8h b_lo = *reinterpret_cast<const v8h*>(brow);
        v8h b_hi = *reinterpret_cast<const v8h*>(brow + 16);
        v16h bfrag = __builtin_shufflevector(b_lo, b_hi,
            0, 1, 2, 3, 4, 5, 6, 7, 8, 9, 10, 11, 12, 13, 14, 15);

        acc = __builtin_amdgcn_wmma_f32_16x16x32_f16(
            /*neg_a=*/false, afrag, /*neg_b=*/false, bfrag,
            /*c_mod=*/(short)0, acc, /*reuse_a=*/false, /*reuse_b=*/false);
    }

    // Merge partial 64x64 sums across the N-splits with HW f32 atomics.
    float* obase = out + (size_t)b * GRID_HW * GRID_HW;
    #pragma unroll
    for (int r = 0; r < 8; ++r) {
        const int h = th * 16 + r + hi * 8;    // C/D layout: VGPR r, lane half
        const int w = tw * 16 + m;
        unsafeAtomicAdd(&obase[h * GRID_HW + w], acc[r]);
    }
}

__global__ __launch_bounds__(THREADS)
void proj_tanh_kernel(float* __restrict__ out) {
    int i = blockIdx.x * blockDim.x + threadIdx.x;
    if (i < OUT_ELEMS) out[i] = tanhf(out[i]);
}

extern "C" void kernel_launch(void* const* d_in, const int* in_sizes, int n_in,
                              void* d_out, int out_size, void* d_ws, size_t ws_size,
                              hipStream_t stream) {
    const float* xyz = (const float*)d_in[0];   // (8, 8192, 3) f32
    const float* az  = (const float*)d_in[1];   // (8,) f32
    const float* el  = (const float*)d_in[2];   // (8,) f32
    float* out = (float*)d_out;                 // (8, 64, 64) f32

    const int blks = (OUT_ELEMS + THREADS - 1) / THREADS;
    proj_zero_kernel<<<blks, THREADS, 0, stream>>>(out);
    proj_splat_wmma_kernel<<<dim3(BS, NSPLIT), THREADS, 0, stream>>>(xyz, az, el, out);
    proj_tanh_kernel<<<blks, THREADS, 0, stream>>>(out);
}
